// ConvNd_60851096649851
// MI455X (gfx1250) — matrix operand order repaired and checked by reading, not compiled, not run on hardware
//
#include <hip/hip_runtime.h>

typedef __attribute__((ext_vector_type(16))) __bf16 v16bf;
typedef __attribute__((ext_vector_type(8)))  __bf16 v8bf;
typedef __attribute__((ext_vector_type(4)))  __bf16 v4bf;
typedef __attribute__((ext_vector_type(8)))  float  v8f;

#define IMG_H 1024
#define IMG_W 1024
#define NPH   127      // patch grid height
#define NPW   127      // patch grid width
#define OUT_H 508
#define OUT_W 508

// ---------------------------------------------------------------------------
// Prep: split fp32 weight [16][256] into bf16 hi/lo planes (bf16x3 scheme).
// ---------------------------------------------------------------------------
__global__ void wsplit_kernel(const float* __restrict__ w,
                              __bf16* __restrict__ whi,
                              __bf16* __restrict__ wlo) {
    int idx = blockIdx.x * 256 + threadIdx.x;
    if (idx < 16 * 256) {
        float f  = w[idx];
        __bf16 h = (__bf16)f;
        __bf16 l = (__bf16)(f - (float)h);
        whi[idx] = h;
        wlo[idx] = l;
    }
}

// ---------------------------------------------------------------------------
// Main kernel: block = 128 threads (4 wave32).
//   blockIdx.x = jt  (0..7)  : 16 patch columns  j = jt*16 + m
//   blockIdx.y = i0  (0..31) : 4 patch rows      i = i0*4 + wave
//   blockIdx.z = b   (0..31) : batch
// Stages a 40x136 fp32 region as bf16 hi/lo in LDS; each wave runs a
// 16x16x256 GEMM as 8 chunks x 3 v_wmma_f32_16x16x32_bf16 on three
// independent accumulators (bf16x3 product terms), summed at the end.
// ---------------------------------------------------------------------------
__launch_bounds__(128)
__global__ void conv_wmma_kernel(const float* __restrict__ x,
                                 const __bf16* __restrict__ whi,
                                 const __bf16* __restrict__ wlo,
                                 float* __restrict__ out) {
    // 40 rows x 144 cols (136 used, padded to keep 16B alignment per row)
    __shared__ __align__(16) __bf16 a_hi[40 * 144];
    __shared__ __align__(16) __bf16 a_lo[40 * 144];

    const int tid  = threadIdx.x;
    const int wv   = tid >> 5;     // wave id 0..3
    const int lane = tid & 31;
    const int l    = lane & 15;
    const int half = lane >> 4;

    const int jt = blockIdx.x;
    const int i0 = blockIdx.y;
    const int b  = blockIdx.z;

    const int r0 = i0 * 32;        // first input row staged
    const int c0 = jt * 128;       // first input col staged

    const float* xb = x + (size_t)b * (IMG_H * IMG_W);

    // ---- stage input region (40 x 136 floats) as bf16 hi/lo --------------
    for (int t = tid; t < 40 * 34; t += 128) {
        int row = t / 34;
        int c4  = t - row * 34;
        int gr  = r0 + row;      if (gr > IMG_H - 1) gr = IMG_H - 1;   // bottom edge
        int gc  = c0 + c4 * 4;   if (gc > IMG_W - 4) gc = IMG_W - 4;   // right edge
        float4 v = *(const float4*)(xb + (size_t)gr * IMG_W + gc);
        __bf16 h0 = (__bf16)v.x, h1 = (__bf16)v.y,
               h2 = (__bf16)v.z, h3 = (__bf16)v.w;
        v4bf hv = { h0, h1, h2, h3 };
        v4bf lv = { (__bf16)(v.x - (float)h0), (__bf16)(v.y - (float)h1),
                    (__bf16)(v.z - (float)h2), (__bf16)(v.w - (float)h3) };
        *(v4bf*)(a_hi + row * 144 + c4 * 4) = hv;
        *(v4bf*)(a_lo + row * 144 + c4 * 4) = lv;
    }
    __syncthreads();

    // ---- GEMM: M=16 patches, N=16 features, K=256 ------------------------
    v8f acc0 = {};   // Ah * Bh
    v8f acc1 = {};   // Ah * Bl
    v8f acc2 = {};   // Al * Bh
    const int rbase = wv * 8;          // this wave's patch row -> LDS row base

    #pragma unroll
    for (int c = 0; c < 8; ++c) {
        // A fragment (16-bit 16x32 layout): lane holds row M=l;
        // lanes 0-15 take K {0-7,16-23}, lanes 16-31 take K {8-15,24-31}.
        // k = wr*16 + wc  -> window rows 2c, 2c+1 ; cols half*8 .. half*8+7.
        const __bf16* pa0h = a_hi + (rbase + 2 * c    ) * 144 + l * 8 + half * 8;
        const __bf16* pa1h = a_hi + (rbase + 2 * c + 1) * 144 + l * 8 + half * 8;
        const __bf16* pa0l = a_lo + (rbase + 2 * c    ) * 144 + l * 8 + half * 8;
        const __bf16* pa1l = a_lo + (rbase + 2 * c + 1) * 144 + l * 8 + half * 8;
        v16bf ah, al;
        ((v8bf*)&ah)[0] = *(const v8bf*)pa0h;
        ((v8bf*)&ah)[1] = *(const v8bf*)pa1h;
        ((v8bf*)&al)[0] = *(const v8bf*)pa0l;
        ((v8bf*)&al)[1] = *(const v8bf*)pa1l;

        // B fragment (32x16): lane = column N=l; lanes 0-15 K {0-7,16-23},
        // lanes 16-31 K {8-15,24-31} of weight row l.
        const __bf16* pwh = whi + l * 256 + c * 32 + half * 8;
        const __bf16* pwl = wlo + l * 256 + c * 32 + half * 8;
        v16bf bh, bl;
        ((v8bf*)&bh)[0] = *(const v8bf*)(pwh);
        ((v8bf*)&bh)[1] = *(const v8bf*)(pwh + 16);
        ((v8bf*)&bl)[0] = *(const v8bf*)(pwl);
        ((v8bf*)&bl)[1] = *(const v8bf*)(pwl + 16);

        // bf16x3: (Ah+Al)(Bh+Bl) ~= AhBh + AhBl + AlBh  (independent accs)
        acc0 = __builtin_amdgcn_wmma_f32_16x16x32_bf16(false, ah, false, bh,
                                                       (short)0, acc0, false, false);
        acc1 = __builtin_amdgcn_wmma_f32_16x16x32_bf16(false, ah, false, bl,
                                                       (short)0, acc1, false, false);
        acc2 = __builtin_amdgcn_wmma_f32_16x16x32_bf16(false, al, false, bh,
                                                       (short)0, acc2, false, false);
    }

    v8f acc = (acc1 + acc2) + acc0;   // small terms first, then dominant term

    // ---- dilated 4x4 scatter --------------------------------------------
    // D layout: element v -> M = v + half*8 (patch within tile), N = l.
    // out[b, i*4 + n/4, j*4 + n%4] = y[b,i,j,n]
    const int i = i0 * 4 + wv;
    if (i < NPH) {
        float* op = out + (size_t)b * (OUT_H * OUT_W)
                  + (i * 4 + (l >> 2)) * OUT_W + (l & 3);
        #pragma unroll
        for (int v = 0; v < 8; ++v) {
            int j = jt * 16 + half * 8 + v;
            if (j < NPW) op[j * 4] = acc[v];
        }
    }
}

extern "C" void kernel_launch(void* const* d_in, const int* in_sizes, int n_in,
                              void* d_out, int out_size, void* d_ws, size_t ws_size,
                              hipStream_t stream) {
    const float* x = (const float*)d_in[0];       // [32,1024,1024] fp32
    const float* w = (const float*)d_in[1];       // [16,256] fp32
    float* out = (float*)d_out;                   // [32,508,508] fp32

    __bf16* whi = (__bf16*)d_ws;                  // 4096 bf16
    __bf16* wlo = whi + 16 * 256;                 // 4096 bf16

    wsplit_kernel<<<16, 256, 0, stream>>>(w, whi, wlo);

    dim3 grid(8, 32, 32);                          // jt, i0, batch
    conv_wmma_kernel<<<grid, 128, 0, stream>>>(x, whi, wlo, out);
}
